// PCT_25855703122107
// MI455X (gfx1250) — matrix operand order repaired and checked by reading, not compiled, not run on hardware
//
#include <hip/hip_runtime.h>
#include <hip/hip_bf16.h>

// ---------------------------------------------------------------------------
// PCT forward for MI455X (gfx1250, wave32). All GEMM-shaped einsums run on
// v_wmma_f32_16x16x32_f16 (one wave = one 16x16 f32 tile, K-step 32).
// A operand: 4x global_load_b128 per K-step (unit K-stride everywhere but the
// energy GEMM). B operand: one running base pointer + 16 hoisted per-lane
// int row offsets -> global_load_b32 (no alloca'd pointer array, keeps
// address-space inference), with global_prefetch of the next tile.
// K%32==0 everywhere except the 3-channel input layer (branch-free tail).
// Irregular stages (FPS / KNN / softmax / pooling) are plain VALU kernels.
// Working set ~150MB -> resident in the 192MB L2; compute-bound design.
// ---------------------------------------------------------------------------

typedef __attribute__((ext_vector_type(16))) _Float16 v16h;
typedef __attribute__((ext_vector_type(8)))  float    v8f;

#define BN_INV 0.99999500003749969f   // (1+1e-5)^-0.5

struct GemmP {
    const float* A;  int a_row; int a_col; long long a_batch;
    const float* B1; const float* B2; int b_row; int b_col; long long b_batch;
    const int* gnb; const int* gct; int dsplit;   // MODE 1 (SG gather) only
    float* O; int o_row; int o_col; long long o_batch;
    const float* pre_b;   // bias added before BN scaling (per output row m)
    const float* scale;   // BN gamma (multiplied by BN_INV in-kernel)
    const float* post_b;  // BN beta
    const float* res;     // residual, indexed with output strides
    int M, N, K;
    int act;              // 0 none, 1 relu, 2 leaky-relu(0.2)
    int pool;             // 1 => atomic-max over groups of 32 output columns
    int n_off;            // column offset added before output indexing
};

__device__ __forceinline__ v16h cvt16(const float* t) {
    v16h r;
#pragma unroll
    for (int e = 0; e < 16; ++e) r[e] = (_Float16)t[e];
    return r;
}

// A fragment, unit K-stride: two contiguous 8-float runs -> 4x b128.
__device__ __forceinline__ v16h load_a16(const float* ap) {
    const float4 a0 = *(const float4*)(ap);
    const float4 a1 = *(const float4*)(ap + 4);
    const float4 a2 = *(const float4*)(ap + 16);
    const float4 a3 = *(const float4*)(ap + 20);
    const float t[16] = {a0.x,a0.y,a0.z,a0.w, a1.x,a1.y,a1.z,a1.w,
                         a2.x,a2.y,a2.z,a2.w, a3.x,a3.y,a3.z,a3.w};
    return cvt16(t);
}

// A fragment, strided K (energy GEMM only): hoisted element offsets.
__device__ __forceinline__ v16h load_a16s(const float* ap, const int* eao) {
    float t[16];
#pragma unroll
    for (int e = 0; e < 16; ++e) t[e] = ap[eao[e]];
    return cvt16(t);
}

// One wave computes a 16x16 output tile.
// MODE 0: B(k,n) = B1[k,n] (- B2[k,n] if HASB2; B2 shares B1's strides).
// MODE 1: SG grouped operand: k<dsplit -> feat[k][nb[n]] - feat[k][ct[n]],
//         k>=dsplit -> feat[k-dsplit][ct[n]]  (concat [norm, center]).
//         dsplit is a multiple of 32 -> two sequential K loops, no branches.
template<int MODE, int HASB2, int ACOL1>
__global__ __launch_bounds__(256) void wmma_gemm(GemmP p) {
    const int wave = threadIdx.x >> 5;
    const int lane = threadIdx.x & 31;
    const int half = lane >> 4;           // 0: lanes 0-15, 1: lanes 16-31
    const int l15  = lane & 15;
    const int tiles_m = p.M >> 4;
    const int tiles_n = p.N >> 4;
    const int tile = blockIdx.x * 8 + wave;
    if (tile >= tiles_m * tiles_n) return;   // wave-uniform; EXEC stays all-1
    const int tm = (tile % tiles_m) << 4;
    const int tn = (tile / tiles_m) << 4;
    const long long zb = blockIdx.z;
    const size_t brow  = (size_t)p.b_row;
    const size_t kstep = brow * 32;

    const int am = tm + l15;   // A row this lane supplies (both lane halves)
    const int bn = tn + l15;   // B column this lane supplies

    const float* Arow = p.A + zb * p.a_batch + (size_t)am * p.a_row;
    const float* B1b  = p.B1 + zb * p.b_batch;

    // Hoisted per-lane B-fragment row offsets (row (e<8?e:e+8)+8*half of the K-step).
    int ebo[16];
#pragma unroll
    for (int e = 0; e < 16; ++e) ebo[e] = ((e < 8 ? e : e + 8) + half * 8) * p.b_row;

    // Running base pointers (advance 32 rows per K-step).
    const float* bbase  = nullptr;   // MODE0: column base; MODE1: center col
    const float* b2base = nullptr;   // MODE0+HASB2
    const float* nbase  = nullptr;   // MODE1: neighbor column
    int cb = 0;                      // column offset for the tail path
    int ctcol = 0;                   // MODE1 center column (for re-basing)
    if (MODE == 0) {
        cb = bn * p.b_col;
        bbase = B1b + cb;
        if (HASB2) b2base = p.B2 + zb * p.b_batch + cb;
    } else {
        ctcol = p.gct[bn];           // hoisted: invariant over K
        bbase = B1b + ctcol;
        nbase = B1b + p.gnb[bn];
    }

    const float* ap = Arow + (ACOL1 ? half * 8 : 0);
    int eao[16];
    if (!ACOL1) {
#pragma unroll
        for (int e = 0; e < 16; ++e)
            eao[e] = ((e < 8 ? e : e + 8) + half * 8) * p.a_col;
    }

    v8f acc = {};
    const int kmain = p.K & ~31;
    int k0 = 0;

    if (MODE == 1) {
        // ---- norm rows: feat[nb] - feat[ct] ----
        for (; k0 < p.dsplit; k0 += 32) {
            const v16h af = load_a16(ap);        // MODE1 A is always unit-stride
            float tb[16], tc[16];
#pragma unroll
            for (int e = 0; e < 16; ++e) tb[e] = nbase[ebo[e]];
#pragma unroll
            for (int e = 0; e < 16; ++e) tc[e] = bbase[ebo[e]];
#pragma unroll
            for (int e = 0; e < 16; ++e) tb[e] -= tc[e];
            acc = __builtin_amdgcn_wmma_f32_16x16x32_f16(
                      false, af, false, cvt16(tb), (short)0, acc, false, false);
            bbase += kstep; nbase += kstep; ap += 32;
            __builtin_prefetch(bbase + ebo[0], 0, 3);
            __builtin_prefetch(nbase + ebo[0], 0, 3);
        }
        // ---- center rows: feat[k-dsplit][ct] -> re-base the row pointer ----
        bbase = B1b + ctcol;
        for (; k0 < p.K; k0 += 32) {
            const v16h af = load_a16(ap);
            float tb[16];
#pragma unroll
            for (int e = 0; e < 16; ++e) tb[e] = bbase[ebo[e]];
            acc = __builtin_amdgcn_wmma_f32_16x16x32_f16(
                      false, af, false, cvt16(tb), (short)0, acc, false, false);
            bbase += kstep; ap += 32;
            __builtin_prefetch(bbase + ebo[0], 0, 3);
            __builtin_prefetch(bbase + ebo[8], 0, 3);
        }
    } else {
        for (; k0 < kmain; k0 += 32) {
            const v16h af = ACOL1 ? load_a16(ap) : load_a16s(ap, eao);
            float tb[16];
#pragma unroll
            for (int e = 0; e < 16; ++e) tb[e] = bbase[ebo[e]];
            if (HASB2) {
                float t2[16];
#pragma unroll
                for (int e = 0; e < 16; ++e) t2[e] = b2base[ebo[e]];
#pragma unroll
                for (int e = 0; e < 16; ++e) tb[e] -= t2[e];
            }
            acc = __builtin_amdgcn_wmma_f32_16x16x32_f16(
                      false, af, false, cvt16(tb), (short)0, acc, false, false);
            bbase += kstep; ap += ACOL1 ? 32 : 32 * p.a_col;
            if (HASB2) b2base += kstep;
            __builtin_prefetch(bbase + ebo[0], 0, 3);
            __builtin_prefetch(bbase + ebo[8], 0, 3);
        }
        // ---- branch-free tail for K%32 != 0 (only the 3-channel input
        //      layer). Clamped index + select -> v_cndmask, no EXEC writes.
        if (k0 < p.K) {
            float ta[16], tb[16];
#pragma unroll
            for (int e = 0; e < 16; ++e) {
                const int ka  = k0 + (e < 8 ? e : e + 8) + half * 8;
                const int kac = ka < p.K ? ka : p.K - 1;
                const float av = ACOL1 ? Arow[kac] : Arow[(size_t)kac * p.a_col];
                ta[e] = ka < p.K ? av : 0.f;
                const int kb  = k0 + (e < 8 ? e : e + 8) + half * 8;
                const int kbc = kb < p.K ? kb : p.K - 1;
                float bv = B1b[(size_t)kbc * brow + cb];
                if (HASB2) bv -= p.B2[zb * p.b_batch + (size_t)kbc * brow + cb];
                tb[e] = kb < p.K ? bv : 0.f;
            }
            acc = __builtin_amdgcn_wmma_f32_16x16x32_f16(
                      false, cvt16(ta), false, cvt16(tb), (short)0, acc, false, false);
        }
    }

    float* Ob = p.O + zb * p.o_batch;
    const float* Rb = p.res ? p.res + zb * p.o_batch : nullptr;
    const int on = tn + l15 + p.n_off;
#pragma unroll
    for (int r = 0; r < 8; ++r) {
        const int m = tm + r + half * 8;   // documented C/D layout
        float v = acc[r];
        if (p.pre_b)  v += p.pre_b[m];
        if (p.scale)  v = p.scale[m] * BN_INV * v + (p.post_b ? p.post_b[m] : 0.f);
        if (p.act == 1)      v = v > 0.f ? v : 0.f;
        else if (p.act == 2) v = v > 0.f ? v : 0.2f * v;
        if (Rb) v += Rb[(size_t)m * p.o_row + (size_t)on * p.o_col];
        if (p.pool) {
            // K=32 group max-pool; post-ReLU v>=0 and dest pre-zeroed,
            // so int-compare atomicMax is exact for these floats.
            const int q = on >> 5;
            atomicMax((int*)&Ob[(size_t)m * p.o_row + (size_t)q * p.o_col],
                      __float_as_int(v));
        } else {
            Ob[(size_t)m * p.o_row + (size_t)on * p.o_col] = v;
        }
    }
}

// ---------------------------------------------------------------------------
// Furthest point sampling: one block per batch, dmin kept in LDS.
// coords layout [B][3][N]. Tie-break = lowest index (matches jnp.argmax).
__global__ void fps_kernel(const float* coords, int N, int s, int* fidx) {
    extern __shared__ float smem[];           // dmin[N] | rv[256] | ri[256]
    float* dmin = smem;
    float* rv = smem + N;
    int*   ri = (int*)(rv + 256);
    __shared__ int cur;
    const int b = blockIdx.x;
    const float* px = coords + (size_t)b * 3 * N;
    const float* py = px + N;
    const float* pz = py + N;
    for (int n = threadIdx.x; n < N; n += 256) dmin[n] = 1e10f;
    if (threadIdx.x == 0) cur = 0;
    __syncthreads();
    for (int it = 0; it < s; ++it) {
        const int far = cur;
        if (threadIdx.x == 0) fidx[b * s + it] = far;
        const float cx = px[far], cy = py[far], cz = pz[far];
        float best = -1.f; int bi = 0x7fffffff;
        for (int n = threadIdx.x; n < N; n += 256) {
            const float dx = px[n]-cx, dy = py[n]-cy, dz = pz[n]-cz;
            const float d = dx*dx + dy*dy + dz*dz;
            const float dm = fminf(dmin[n], d);
            dmin[n] = dm;
            if (dm > best) { best = dm; bi = n; }
        }
        rv[threadIdx.x] = best; ri[threadIdx.x] = bi;
        __syncthreads();
        for (int off = 128; off > 0; off >>= 1) {
            if (threadIdx.x < off) {
                const float ov = rv[threadIdx.x + off];
                const int   oi = ri[threadIdx.x + off];
                if (ov > rv[threadIdx.x] ||
                    (ov == rv[threadIdx.x] && oi < ri[threadIdx.x])) {
                    rv[threadIdx.x] = ov; ri[threadIdx.x] = oi;
                }
            }
            __syncthreads();
        }
        if (threadIdx.x == 0) cur = ri[0];
        __syncthreads();
    }
}

// KNN(32): one thread per query keeps a sorted top-32; emits global point
// indices (b*N+idx) for neighbor and center, ready for the gather-GEMM.
__global__ void knn_kernel(const float* coords, int N, int s, int total,
                           const int* fidx, int* nb_idx, int* ct_idx) {
    const int t = blockIdx.x * blockDim.x + threadIdx.x;
    if (t >= total) return;
    const int b = t / s;
    const float* px = coords + (size_t)b * 3 * N;
    const float* py = px + N;
    const float* pz = py + N;
    const int ctr = fidx[t];
    const float qx = px[ctr], qy = py[ctr], qz = pz[ctr];
    float db[32]; int ib[32];
#pragma unroll
    for (int i = 0; i < 32; ++i) { db[i] = 3.0e38f; ib[i] = 0; }
    for (int n = 0; n < N; ++n) {
        const float dx = px[n]-qx, dy = py[n]-qy, dz = pz[n]-qz;
        const float d = dx*dx + dy*dy + dz*dz;
        if (d < db[31]) {
            int j = 31;
            while (j > 0 && db[j-1] > d) { db[j] = db[j-1]; ib[j] = ib[j-1]; --j; }
            db[j] = d; ib[j] = n;
        }
    }
    for (int i = 0; i < 32; ++i) {
        nb_idx[t * 32 + i] = b * N + ib[i];
        ct_idx[t * 32 + i] = b * N + ctr;
    }
}

// xyz1[b][c][i] = coords[b][c][fidx[b*s+i]]
__global__ void gather_xyz_kernel(const float* coords, int N, int s,
                                  const int* fidx, float* out, int total) {
    const int t = blockIdx.x * blockDim.x + threadIdx.x;
    if (t >= total) return;
    const int b = t / (3 * s);
    const int c = (t / s) % 3;
    const int i = t % s;
    out[t] = coords[(size_t)b * 3 * N + (size_t)c * N + fidx[b * s + i]];
}

// Row softmax over the last axis of energy [B*256][256] (in place).
__global__ void softmax_rows_kernel(float* e) {
    __shared__ float red[256];
    const int t = threadIdx.x;
    float* p = e + (size_t)blockIdx.x * 256;
    const float v = p[t];
    red[t] = v; __syncthreads();
    for (int o = 128; o > 0; o >>= 1) {
        if (t < o) red[t] = fmaxf(red[t], red[t + o]);
        __syncthreads();
    }
    const float mx = red[0]; __syncthreads();
    const float ex = __expf(v - mx);
    red[t] = ex; __syncthreads();
    for (int o = 128; o > 0; o >>= 1) {
        if (t < o) red[t] += red[t + o];
        __syncthreads();
    }
    p[t] = ex / red[0];
}

// attn[b,n,m] /= (1e-9 + sum_n attn[b,n,m]) — one block per (b,m) column.
__global__ void colnorm_kernel(float* attn) {
    __shared__ float red[256];
    const int b = blockIdx.x >> 8;
    const int m = blockIdx.x & 255;
    float* base = attn + (size_t)b * 65536 + m;
    const int t = threadIdx.x;
    const float v = base[(size_t)t * 256];
    red[t] = v; __syncthreads();
    for (int o = 128; o > 0; o >>= 1) {
        if (t < o) red[t] += red[t + o];
        __syncthreads();
    }
    base[(size_t)t * 256] = v / (1e-9f + red[0]);
}

// hmax[b][c] = max over 256 points of hl[c][b*256+pt]   (hl is [1024][4096])
__global__ void maxpts_kernel(const float* hl, float* out) {
    const int t = blockIdx.x * blockDim.x + threadIdx.x;
    if (t >= 16 * 1024) return;
    const int b = t / 1024, c = t % 1024;
    const float* p = hl + (size_t)c * 4096 + b * 256;
    float m = -3.0e38f;
    for (int i = 0; i < 256; ++i) m = fmaxf(m, p[i]);
    out[b * 1024 + c] = m;
}

// ---------------------------------------------------------------------------
extern "C" void kernel_launch(void* const* d_in, const int* in_sizes, int n_in,
                              void* d_out, int out_size, void* d_ws, size_t ws_size,
                              hipStream_t stream) {
    (void)in_sizes; (void)n_in; (void)out_size; (void)ws_size;
    const float* x = (const float*)d_in[0];          // [16][3][4096]
    auto P = [&](int i) { return (const float*)d_in[i]; };

    // ---- workspace carve (≈150 MB total; L2 resident) ----
    char* wp = (char*)d_ws;
    auto alloc = [&](size_t nfloats) {
        float* r = (float*)wp; wp += nfloats * sizeof(float); return r;
    };
    float* fA     = alloc((size_t)64 * 65536);    // NE hidden  [64][B*4096]
    float* fB     = alloc((size_t)64 * 65536);    // NE out     [64][B*4096]
    float* h1     = alloc((size_t)256 * 65536);   // SG conv1 chunk buffer
    float* f1     = alloc((size_t)128 * 8192);    // SG1 pooled [128][B*512]
    float* xyz1   = alloc((size_t)16 * 3 * 512);  // [B][3][512]
    float* cat    = alloc((size_t)1280 * 4096);   // concat [1280][B*256]
    float* qb     = alloc((size_t)64 * 4096);     // OA q/k    [64][B*256]
    float* vb     = alloc((size_t)256 * 4096);    // OA v      [256][B*256]
    float* energy = alloc((size_t)16 * 256 * 256);
    float* xrb    = alloc((size_t)256 * 4096);
    float* hl     = alloc((size_t)1024 * 4096);
    float* hmax   = alloc((size_t)16 * 1024);
    float* hfc1   = alloc((size_t)512 * 16);
    int* fidx1 = (int*)alloc(16 * 512);
    int* fidx2 = (int*)alloc(16 * 256);
    int* nbIdx = (int*)alloc((size_t)16 * 512 * 32);
    int* ctIdx = (int*)alloc((size_t)16 * 512 * 32);

    auto gemm = [&](int mode, int M, int N, int K, int nb,
                    const float* A, int ar, int ac, long long ab,
                    const float* B1, const float* B2, int br, int bc, long long bbat,
                    const int* gn, const int* gc, int ds,
                    float* O, int orr, int oc, long long ob,
                    const float* pre, const float* sc, const float* po,
                    const float* res, int act, int pool, int noff) {
        GemmP p{A, ar, ac, ab, B1, B2, br, bc, bbat, gn, gc, ds,
                O, orr, oc, ob, pre, sc, po, res, M, N, K, act, pool, noff};
        const int tiles = (M / 16) * (N / 16);
        dim3 grid((tiles + 7) / 8, 1, nb);
        if (mode == 1)      wmma_gemm<1, 0, 1><<<grid, 256, 0, stream>>>(p);
        else if (B2)        wmma_gemm<0, 1, 1><<<grid, 256, 0, stream>>>(p);
        else if (ac == 1)   wmma_gemm<0, 0, 1><<<grid, 256, 0, stream>>>(p);
        else                wmma_gemm<0, 0, 0><<<grid, 256, 0, stream>>>(p);
    };

    // ---- NeighborEmbedding pointwise MLP: 3->64->64 ----
    gemm(0, 64, 4096, 3, 16,
         P(1), 3, 1, 0,                 // ne_w1 [64][3]
         x, nullptr, 4096, 1, 12288,    // x batched [B][3][4096]
         nullptr, nullptr, 0,
         fA, 65536, 1, 4096,            // fA flat [64][B*4096]
         nullptr, P(2), P(3), nullptr, 1, 0, 0);
    gemm(0, 64, 65536, 64, 1,
         P(4), 64, 1, 0,
         fA, nullptr, 65536, 1, 0,
         nullptr, nullptr, 0,
         fB, 65536, 1, 0,
         nullptr, P(5), P(6), nullptr, 1, 0, 0);

    // ---- SG block 1: 4096 -> 512, D=64 -> C=128 ----
    fps_kernel<<<16, 256, (4096 + 512) * sizeof(float), stream>>>(x, 4096, 512, fidx1);
    knn_kernel<<<32, 256, 0, stream>>>(x, 4096, 512, 16 * 512, fidx1, nbIdx, ctIdx);
    gather_xyz_kernel<<<96, 256, 0, stream>>>(x, 4096, 512, fidx1, xyz1, 16 * 3 * 512);
    hipMemsetAsync(f1, 0, (size_t)128 * 8192 * sizeof(float), stream);
    for (int ch = 0; ch < 4; ++ch) {                    // 262144 cols in 4 chunks
        const int c0 = ch * 65536;
        gemm(1, 128, 65536, 128, 1,
             P(7), 128, 1, 0,                            // sg1_w1
             fB, nullptr, 65536, 0, 0,                   // feats [64][B*4096]
             nbIdx + c0, ctIdx + c0, 64,
             h1, 65536, 1, 0,
             nullptr, P(8), P(9), nullptr, 1, 0, 0);
        gemm(0, 128, 65536, 128, 1,
             P(10), 128, 1, 0,                           // sg1_w2
             h1, nullptr, 65536, 1, 0,
             nullptr, nullptr, 0,
             f1, 8192, 1, 0,                             // pooled [128][B*512]
             nullptr, P(11), P(12), nullptr, 1, 1, c0);
    }

    // ---- SG block 2: 512 -> 256, D=128 -> C=256 (pooled straight into cat) ----
    fps_kernel<<<16, 256, (512 + 512) * sizeof(float), stream>>>(xyz1, 512, 256, fidx2);
    knn_kernel<<<16, 256, 0, stream>>>(xyz1, 512, 256, 16 * 256, fidx2, nbIdx, ctIdx);
    hipMemsetAsync(cat, 0, (size_t)256 * 4096 * sizeof(float), stream);
    for (int ch = 0; ch < 2; ++ch) {                     // 131072 cols in 2 chunks
        const int c0 = ch * 65536;
        gemm(1, 256, 65536, 256, 1,
             P(13), 256, 1, 0,                           // sg2_w1
             f1, nullptr, 8192, 0, 0,                    // feats [128][B*512]
             nbIdx + c0, ctIdx + c0, 128,
             h1, 65536, 1, 0,
             nullptr, P(14), P(15), nullptr, 1, 0, 0);
        gemm(0, 256, 65536, 256, 1,
             P(16), 256, 1, 0,                           // sg2_w2
             h1, nullptr, 65536, 1, 0,
             nullptr, nullptr, 0,
             cat, 4096, 1, 0,                            // h0 rows 0..255 of cat
             nullptr, P(17), P(18), nullptr, 1, 1, c0);
    }

    // ---- 4x Offset-Attention; each output written to the next cat slab ----
    for (int i = 0; i < 4; ++i) {
        const int pb = 19 + i * 7;   // qkw, vw, vb, tw, tb, g, b
        float* xin  = cat + (size_t)i * 256 * 4096;
        float* xout = cat + (size_t)(i + 1) * 256 * 4096;
        gemm(0, 64, 4096, 256, 1,                        // q = k = qkw @ x
             P(pb), 256, 1, 0,
             xin, nullptr, 4096, 1, 0,
             nullptr, nullptr, 0,
             qb, 4096, 1, 0,
             nullptr, nullptr, nullptr, nullptr, 0, 0, 0);
        gemm(0, 256, 4096, 256, 1,                       // v = vw @ x + vb
             P(pb + 1), 256, 1, 0,
             xin, nullptr, 4096, 1, 0,
             nullptr, nullptr, 0,
             vb, 4096, 1, 0,
             P(pb + 2), nullptr, nullptr, nullptr, 0, 0, 0);
        gemm(0, 256, 256, 64, 16,                        // energy[b] = q_b^T q_b
             qb, 1, 4096, 256,
             qb, nullptr, 4096, 1, 256,
             nullptr, nullptr, 0,
             energy, 256, 1, 65536,
             nullptr, nullptr, nullptr, nullptr, 0, 0, 0);
        softmax_rows_kernel<<<4096, 256, 0, stream>>>(energy);
        colnorm_kernel<<<4096, 256, 0, stream>>>(energy);
        gemm(0, 256, 256, 256, 16,                       // xr[b] = v_b @ attn_b
             vb, 4096, 1, 256,
             energy, nullptr, 256, 1, 65536,
             nullptr, nullptr, 0,
             xrb, 4096, 1, 256,
             nullptr, nullptr, nullptr, nullptr, 0, 0, 0);
        gemm(0, 256, 4096, 256, 1,                       // x + relu(bn(tw(x-xr)+tb))
             P(pb + 3), 256, 1, 0,
             xin, xrb, 4096, 1, 0,
             nullptr, nullptr, 0,
             xout, 4096, 1, 0,
             P(pb + 4), P(pb + 5), P(pb + 6), xin, 1, 0, 0);
    }

    // ---- lin 1280->1024, leaky-relu, max over points ----
    gemm(0, 1024, 4096, 1280, 1,
         P(47), 1280, 1, 0,
         cat, nullptr, 4096, 1, 0,
         nullptr, nullptr, 0,
         hl, 4096, 1, 0,
         nullptr, P(48), P(49), nullptr, 2, 0, 0);
    maxpts_kernel<<<64, 256, 0, stream>>>(hl, hmax);

    // ---- FC head: 1024->512 (bn,relu) -> 256 (+bias, bn, relu) ----
    gemm(0, 512, 16, 1024, 1,
         P(50), 1024, 1, 0,
         hmax, nullptr, 1, 1024, 0,
         nullptr, nullptr, 0,
         hfc1, 16, 1, 0,
         nullptr, P(51), P(52), nullptr, 1, 0, 0);
    gemm(0, 256, 16, 512, 1,
         P(53), 512, 1, 0,
         hfc1, nullptr, 16, 1, 0,
         nullptr, nullptr, 0,
         (float*)d_out, 1, 256, 0,                       // d_out[b*256+c]
         P(54), P(55), P(56), nullptr, 1, 0, 0);
}
